// TriParser_82068235092240
// MI455X (gfx1250) — matrix operand order repaired and checked by reading, not compiled, not run on hardware
//
#include <hip/hip_runtime.h>
#include <hip/hip_bf16.h>

typedef __attribute__((ext_vector_type(16))) __bf16 v16bf;
typedef __attribute__((ext_vector_type(8)))  __bf16 v8bf;
typedef __attribute__((ext_vector_type(8)))  float  v8f;
typedef __attribute__((ext_vector_type(4)))  float  v4f;
typedef unsigned int u32x4 __attribute__((ext_vector_type(4)));
typedef int          i32x8 __attribute__((ext_vector_type(8)));
typedef int          i32x4 __attribute__((ext_vector_type(4)));

static constexpr int Bq = 8;
static constexpr int Sq = 127;
static constexpr int Nq = 128;          // seq with sentinel
static constexpr int Eq = 512;
static constexpr int Aq = 256;
static constexpr int MTOT = Bq * Nq;    // 1024
static constexpr int NCD = Aq * Aq;     // 65536

#if defined(__has_builtin)
#if __has_builtin(__builtin_amdgcn_tensor_load_to_lds) && \
    __has_builtin(__builtin_amdgcn_s_wait_tensorcnt)
#define HAVE_TDM 1
#endif
#endif

#ifdef HAVE_TDM
#warning "CDNA5 probe: tensor_load_to_lds builtin AVAILABLE - TDM path compiled in"
#else
#warning "CDNA5 probe: tensor_load_to_lds builtin NOT available - manual copy fallback"
#endif

// ---------------- WMMA fragment loaders (contiguous k = 16*(l/16)+e, same for A and B) ----------------
// A fragment 16x32 bf16: lane l holds row m=l%16; g=l/16 selects K-halves.
__device__ __forceinline__ v16bf load_frag_a(const __bf16* base, int ld, int lane) {
  const int m = lane & 15;
  const int g = lane >> 4;
  const __bf16* row = base + m * ld;
  v16bf a;
#pragma unroll
  for (int e = 0; e < 16; ++e) {
    int k = g * 16 + e;
    a[e] = row[k];
  }
  return a;
}

// B fragment 32x16 bf16 from LDS stored [n][K]: lane l holds column n=l%16,
// h=l/16 selects K half.
__device__ __forceinline__ v16bf load_frag_b(const __bf16* base, int ld, int lane) {
  const int n = lane & 15;
  const int h = lane >> 4;
  const __bf16* row = base + n * ld;
  v16bf b;
#pragma unroll
  for (int e = 0; e < 16; ++e) b[e] = row[h * 16 + e];
  return b;
}

__device__ __forceinline__ v8f wmma_bf16(v16bf a, v16bf b, v8f c) {
  return __builtin_amdgcn_wmma_f32_16x16x32_bf16(false, a, false, b, (short)0, c,
                                                 false, false);
}

// ------------- TDM: 2D bf16 tile  global -> LDS  (ISA ch.8 D# layout) -------
#ifdef HAVE_TDM
__device__ __forceinline__ void tdm_load_2d_bf16(unsigned lds_off, const void* gptr,
                                                 unsigned rows, unsigned cols,
                                                 unsigned stride_elems) {
  unsigned long long ga = (unsigned long long)(__UINTPTR_TYPE__)gptr;
  u32x4 g0;
  g0[0] = 1u;                                              // count=1, user mode
  g0[1] = lds_off;                                         // lds_addr (bytes)
  g0[2] = (unsigned)(ga & 0xffffffffull);                  // global_addr[31:0]
  g0[3] = (unsigned)((ga >> 32) & 0x1ffffffull) | (2u << 30);  // addr hi | type=2
  i32x8 g1;
  g1[0] = (int)(1u << 16);                                 // wg_mask=0, data_size=2B
  g1[1] = (int)((cols & 0xffffu) << 16);                   // tensor_dim0[15:0]
  g1[2] = (int)(((cols >> 16) & 0xffffu) | ((rows & 0xffffu) << 16)); // d0 hi|d1 lo
  g1[3] = (int)(((rows >> 16) & 0xffffu) | ((cols & 0xffffu) << 16)); // d1 hi|tile_dim0
  g1[4] = (int)(rows & 0xffffu);                           // tile_dim1, tile_dim2=0
  g1[5] = (int)stride_elems;                               // tensor_dim0_stride lo
  g1[6] = 0;
  g1[7] = 0;
  i32x4 z4 = {0, 0, 0, 0};
#if __clang_major__ >= 23
  i32x8 z8 = {0, 0, 0, 0, 0, 0, 0, 0};
  __builtin_amdgcn_tensor_load_to_lds(g0, g1, z4, z4, z8, 0);
#else
  __builtin_amdgcn_tensor_load_to_lds(g0, g1, z4, z4, 0);
#endif
}
#endif

// ---------------- Kernel A: 7 projections, ELU, fp32 -> bf16 ----------------
struct ProjArgs {
  const float* W[7];
  const float* bias[7];
};

__global__ __launch_bounds__(256) void proj_elu_kernel(
    const float* __restrict__ x, const float* __restrict__ sentinel,
    ProjArgs pa, __bf16* __restrict__ proj /* [7][1024][256] */) {
  const int br = blockIdx.z;
  const int m0 = blockIdx.x * 128;
  const int n0 = blockIdx.y * 128;
  const float* W = pa.W[br];
  const float* bias = pa.bias[br];
  __bf16* out = proj + (size_t)br * MTOT * Aq;

  __shared__ __bf16 lds_a[128 * 32];   // h tile [m][k]
  __shared__ __bf16 lds_b[128 * 32];   // W tile [n][k] (transposed)

  const int tid = threadIdx.x, lane = tid & 31, wave = tid >> 5;
  const v8f vz = {0.f, 0.f, 0.f, 0.f, 0.f, 0.f, 0.f, 0.f};
  v8f acc[8];
#pragma unroll
  for (int i = 0; i < 8; ++i) acc[i] = vz;

  for (int ks = 0; ks < Eq / 32; ++ks) {
    const int k0 = ks * 32;
    __syncthreads();
#pragma unroll
    for (int i = 0; i < 16; ++i) {          // stage A (h = [sentinel; x])
      int idx = tid + i * 256;              // 128x32
      int r = idx >> 5, c = idx & 31;
      int gm = m0 + r, b = gm >> 7, pos = gm & 127;
      float v = (pos == 0) ? sentinel[k0 + c]
                           : x[((size_t)b * Sq + (pos - 1)) * Eq + k0 + c];
      lds_a[idx] = (__bf16)v;
    }
#pragma unroll
    for (int i = 0; i < 16; ++i) {          // stage B: W[k][n] -> lds[n][k]
      int idx = tid + i * 256;              // 32x128
      int kr = idx >> 7, nc = idx & 127;
      lds_b[nc * 32 + kr] = (__bf16)W[(size_t)(k0 + kr) * Aq + n0 + nc];
    }
    __syncthreads();
    v16bf af = load_frag_a(lds_a + wave * 16 * 32, 32, lane);
#pragma unroll
    for (int nt = 0; nt < 8; ++nt) {
      v16bf bf = load_frag_b(lds_b + nt * 16 * 32, 32, lane);
      acc[nt] = wmma_bf16(af, bf, acc[nt]);
    }
  }
  const int nloc = lane & 15, g = lane >> 4;
#pragma unroll
  for (int nt = 0; nt < 8; ++nt) {
    int gn = n0 + nt * 16 + nloc;
    float bv = bias[gn];
#pragma unroll
    for (int p = 0; p < 8; ++p) {
      int gm = m0 + wave * 16 + p + 8 * g;
      float z = acc[nt][p] + bv;
      z = z > 0.f ? z : (__expf(z) - 1.f);   // ELU
      out[(size_t)gm * Aq + gn] = (__bf16)z;
    }
  }
}

// --------- Kernel B: stage-1 GEMM  T1[bi, c*256+d] = u[bi,:] @ U ------------
// 256x128 tile per WG; wave computes 32x128 (each B fragment feeds 2 WMMAs).
// Software-pipelined: next k-step's global loads are issued while WMMAs run.
// Output restaged through per-wave-private LDS for coalesced b128 stores.
__global__ __launch_bounds__(256) void stage1_gemm_kernel(
    const __bf16* __restrict__ u /*[1024][256]*/,
    const float* __restrict__ U /*[256][65536]*/,
    __bf16* __restrict__ T1 /*[1024][65536]*/) {
  const int m0 = blockIdx.x * 256;
  const int n0 = blockIdx.y * 128;
  __shared__ __bf16 lds_a[256 * 32];        // 16KB
  __shared__ __bf16 lds_b[128 * 32];        //  8KB
  __shared__ __bf16 lds_out[8][32 * 128];   // 64KB, per-wave private
  const int tid = threadIdx.x, lane = tid & 31, wave = tid >> 5;
  const v8f vz = {0.f, 0.f, 0.f, 0.f, 0.f, 0.f, 0.f, 0.f};
  v8f acc0[8], acc1[8];
#pragma unroll
  for (int i = 0; i < 8; ++i) { acc0[i] = vz; acc1[i] = vz; }

  __bf16 ra[32];   // register-staged A tile slice (already bf16)
  float rb[16];    // register-staged B tile slice (fp32 from U)

  auto global_stage = [&](int ks) {
    const int k0 = ks * 32;
#pragma unroll
    for (int i = 0; i < 32; ++i) {
      int idx = tid + i * 256;
      ra[i] = u[(size_t)(m0 + (idx >> 5)) * Aq + k0 + (idx & 31)];
    }
#pragma unroll
    for (int i = 0; i < 16; ++i) {
      int idx = tid + i * 256;
      rb[i] = U[(size_t)(k0 + (idx >> 7)) * NCD + n0 + (idx & 127)];
    }
  };
  auto lds_stage = [&]() {
#pragma unroll
    for (int i = 0; i < 32; ++i) lds_a[tid + i * 256] = ra[i];
#pragma unroll
    for (int i = 0; i < 16; ++i) {
      int idx = tid + i * 256;
      lds_b[(idx & 127) * 32 + (idx >> 7)] = (__bf16)rb[i];
    }
  };

  global_stage(0);
  for (int ks = 0; ks < Aq / 32; ++ks) {
    lds_stage();
    __syncthreads();
    if (ks + 1 < Aq / 32) global_stage(ks + 1);  // overlap with WMMA below
    v16bf af0 = load_frag_a(lds_a + (wave * 32) * 32, 32, lane);
    v16bf af1 = load_frag_a(lds_a + (wave * 32 + 16) * 32, 32, lane);
#pragma unroll
    for (int nt = 0; nt < 8; ++nt) {
      v16bf bf = load_frag_b(lds_b + nt * 16 * 32, 32, lane);
      acc0[nt] = wmma_bf16(af0, bf, acc0[nt]);
      acc1[nt] = wmma_bf16(af1, bf, acc1[nt]);
    }
    __syncthreads();
  }

  // Epilogue: restage through per-wave LDS, then coalesced 16B global stores.
  const int nloc = lane & 15, g = lane >> 4;
  __bf16* so = &lds_out[wave][0];            // [32][128]
#pragma unroll
  for (int nt = 0; nt < 8; ++nt) {
#pragma unroll
    for (int p = 0; p < 8; ++p) {
      int r0 = p + 8 * g;
      so[r0 * 128 + nt * 16 + nloc] = (__bf16)acc0[nt][p];
      so[(r0 + 16) * 128 + nt * 16 + nloc] = (__bf16)acc1[nt][p];
    }
  }
  // LDS ops are wave-ordered: no barrier needed for private region readback.
#pragma unroll
  for (int i = 0; i < 16; ++i) {
    int idx = lane * 8 + i * 256;            // 4096 bf16 per wave
    int r = idx >> 7, col = idx & 127;
    v8bf pk = *(const v8bf*)(so + idx);
    *(v8bf*)&T1[(size_t)(m0 + wave * 32 + r) * NCD + n0 + col] = pk;
  }
}

// --- Kernel C: fused stages 2+3, one (b,i) per workgroup, all in 320KB LDS --
//   T2 = V_b(128x256) @ T1[b,i](256x256);  out = T2 @ W_b^T -> [128][128] f32
// LDS byte offsets (dynamic LDS base = 0):
//   sT1t: 0       [d=256][c=256] 128KB   sV: 131072 [j][c] 64KB
//   sW:   196608  [k][d]          64KB   sT2: 262144 [j][d] 64KB
__global__ __launch_bounds__(256, 1) void tri_fused_kernel(
    const __bf16* __restrict__ vproj, const __bf16* __restrict__ wproj,
    const __bf16* __restrict__ T1, float* __restrict__ outbr) {
  const int bi = blockIdx.x;       // b*128 + i
  const int b = bi >> 7;
  extern __shared__ __bf16 smem[];
  __bf16* sT1t = smem;             // [d=256][c=256] (transposed: K=c contiguous)
  __bf16* sV = smem + 65536;
  __bf16* sW = sV + 32768;
  __bf16* sT2 = sW + 32768;
  const int tid = threadIdx.x, lane = tid & 31, wave = tid >> 5;

  const __bf16* gV = vproj + (size_t)b * 128 * 256;
  const __bf16* gW = wproj + (size_t)b * 128 * 256;

#ifdef HAVE_TDM
  // Tensor Data Mover pulls the V_b / W_b tiles into LDS asynchronously while
  // all 8 waves do the T1 transpose below.  One wave issues; TENSORcnt tracks.
  if (wave == 0) {
    tdm_load_2d_bf16(131072u, gV, 128u, 256u, 256u);
    tdm_load_2d_bf16(196608u, gW, 128u, 256u, 256u);
  }
#else
#pragma unroll
  for (int t = 0; t < 16; ++t) {
    int idx = (tid + t * 256) * 8;
    *(v8bf*)(sV + idx) = *(const v8bf*)(gV + idx);
    *(v8bf*)(sW + idx) = *(const v8bf*)(gW + idx);
  }
#endif

  // stage T1[b,i] with transpose ([c][d] global -> [d][c] LDS)
  const __bf16* gT1 = T1 + (size_t)bi * NCD;
  for (int t = 0; t < 32; ++t) {
    int c = (tid >> 5) + t * 8;
    int d0 = (tid & 31) * 8;
    v8bf pk = *(const v8bf*)(gT1 + (size_t)c * 256 + d0);
#pragma unroll
    for (int q = 0; q < 8; ++q) sT1t[(size_t)(d0 + q) * 256 + c] = pk[q];
  }
#ifdef HAVE_TDM
  if (wave == 0) __builtin_amdgcn_s_wait_tensorcnt(0);
#endif
  __syncthreads();

  // Phase 1: T2[j][d] = sum_c V[j][c] * T1t[d][c]
  for (int nt = 0; nt < 16; ++nt) {
    v8f acc = {0.f, 0.f, 0.f, 0.f, 0.f, 0.f, 0.f, 0.f};
#pragma unroll
    for (int kt = 0; kt < 8; ++kt) {
      v16bf af = load_frag_a(sV + (wave * 16) * 256 + kt * 32, 256, lane);
      v16bf bf = load_frag_b(sT1t + (nt * 16) * 256 + kt * 32, 256, lane);
      acc = wmma_bf16(af, bf, acc);
    }
    const int nloc = lane & 15, g = lane >> 4;
    const int d = nt * 16 + nloc;
#pragma unroll
    for (int p = 0; p < 8; ++p)
      sT2[(size_t)(wave * 16 + p + 8 * g) * 256 + d] = (__bf16)acc[p];
  }
  __syncthreads();   // also makes sT1t region dead: reusable below

  // Phase 2: out[j][k] = sum_d T2[j][d] * W[k][d]
  // fp32 result restaged through the (now free) sT1t region, per-wave private,
  // then written back as coalesced 16B stores.
  float* sO = (float*)smem + wave * (16 * 128);   // 8KB per wave inside sT1t
  for (int nt = 0; nt < 8; ++nt) {
    v8f acc = {0.f, 0.f, 0.f, 0.f, 0.f, 0.f, 0.f, 0.f};
#pragma unroll
    for (int kt = 0; kt < 8; ++kt) {
      v16bf af = load_frag_a(sT2 + (wave * 16) * 256 + kt * 32, 256, lane);
      v16bf bf = load_frag_b(sW + (nt * 16) * 256 + kt * 32, 256, lane);
      acc = wmma_bf16(af, bf, acc);
    }
    const int nloc = lane & 15, g = lane >> 4;
    const int kcol = nt * 16 + nloc;
#pragma unroll
    for (int p = 0; p < 8; ++p) sO[(p + 8 * g) * 128 + kcol] = acc[p];
  }
  float* gout = outbr + (size_t)bi * 128 * 128 + (size_t)(wave * 16) * 128;
#pragma unroll
  for (int i = 0; i < 16; ++i) {
    int idx = lane * 4 + i * 128;                 // 2048 f32 per wave
    v4f pk = *(const v4f*)(sO + idx);
    *(v4f*)(gout + idx) = pk;
  }
}

// ---------------- Kernel D: mask_full = [ones; mask] as float ---------------
__global__ void mask_kernel(const int* __restrict__ mask, float* __restrict__ om) {
  int t = blockIdx.x * blockDim.x + threadIdx.x;
  if (t < MTOT) {
    int b = t >> 7, pos = t & 127;
    om[t] = (pos == 0) ? 1.0f : (float)mask[b * Sq + pos - 1];
  }
}

extern "C" void kernel_launch(void* const* d_in, const int* in_sizes, int n_in,
                              void* d_out, int out_size, void* d_ws, size_t ws_size,
                              hipStream_t stream) {
  const float* x = (const float*)d_in[0];
  const int* mask = (const int*)d_in[2];
  ProjArgs pa;
  for (int i = 0; i < 7; ++i) {
    pa.W[i] = (const float*)d_in[3 + 2 * i];
    pa.bias[i] = (const float*)d_in[4 + 2 * i];
  }
  const float* Umats[3] = {(const float*)d_in[17], (const float*)d_in[18],
                           (const float*)d_in[19]};
  const float* sent = (const float*)d_in[20];

  __bf16* proj = (__bf16*)d_ws;                       // 7 * 1024*256 bf16 = 3.5MB
  __bf16* T1 = (__bf16*)((char*)d_ws + (8u << 20));   // 1024*65536 bf16 = 128MB
  float* out = (float*)d_out;
  const size_t branch_elems = (size_t)MTOT * Nq * Nq; // 16,777,216

  // 1) seven ELU projections -> bf16 workspace
  proj_elu_kernel<<<dim3(8, 2, 7), 256, 0, stream>>>(x, sent, pa, proj);

  const __bf16* pr[7];
  for (int i = 0; i < 7; ++i) pr[i] = proj + (size_t)i * MTOT * Aq;
  // branch -> (u, v, w) projection indices, per reference:
  //   sib = tri(sib_head, sib_dep, sib_dep, U_sib)
  //   cop = tri(cop_head, cop_dep, cop_head, U_cop)
  //   gp  = tri(gp_head, gp_head_dep, gp_dep, U_gp)
  const int uu[3] = {0, 2, 4}, vv[3] = {1, 3, 6}, ww[3] = {1, 2, 5};

  for (int br = 0; br < 3; ++br) {
    stage1_gemm_kernel<<<dim3(4, 512), 256, 0, stream>>>(pr[uu[br]], Umats[br], T1);
    tri_fused_kernel<<<dim3(1024), 256, 327680, stream>>>(
        pr[vv[br]], pr[ww[br]], T1, out + (size_t)br * branch_elems);
  }
  mask_kernel<<<dim3(4), 256, 0, stream>>>(mask, out + 3 * branch_elems);
}
